// GraphAttentionEmbedding_69793218560128
// MI455X (gfx1250) — matrix operand order repaired and checked by reading, not compiled, not run on hardware
//
#include <hip/hip_runtime.h>
#include <hip/hip_bf16.h>

// ---------------------------------------------------------------------------
// Problem constants (from the reference)
// ---------------------------------------------------------------------------
#define NN   50000      // nodes
#define NE   400000     // edges
#define IN_CH 128
#define C_CH  32
#define H1_H  8
#define MSG_D 64
#define TD_D  64
#define ED_D  128       // MSG_D + TD_D
#define HID_D 256       // H1_H * C_CH

typedef __attribute__((ext_vector_type(16))) __bf16 v16bf;
typedef __attribute__((ext_vector_type(8)))  float  v8f;

// ---------------------------------------------------------------------------
// Helpers
// ---------------------------------------------------------------------------
__device__ __forceinline__ void atomicMaxF(float* addr, float v) {
    if (!(v == v)) return;                       // NaN guard
    if (v >= 0.0f) atomicMax((int*)addr, __float_as_int(v));
    else           atomicMin((unsigned int*)addr, __float_as_uint(v));
}

// exact bf16(pair packed in u32) -> 2x f32
__device__ __forceinline__ float2 bf16x2_to_f32x2(unsigned u) {
    float2 r;
    r.x = __uint_as_float(u << 16);
    r.y = __uint_as_float(u & 0xFFFF0000u);
    return r;
}

__global__ void f32_to_bf16_kernel(const float* __restrict__ in,
                                   __bf16* __restrict__ out, size_t n) {
    size_t i = (size_t)blockIdx.x * blockDim.x + threadIdx.x;
    if (i < n) out[i] = (__bf16)in[i];
}

__global__ void fill_f32_kernel(float* __restrict__ p, float v, size_t n) {
    size_t i = (size_t)blockIdx.x * blockDim.x + threadIdx.x;
    if (i < n) p[i] = v;
}

// edge_attr = [cos(rel_t * tw + tb) | msg]   -> bf16 (E, 128)
__global__ void edge_attr_kernel(const long long* __restrict__ eidx,
                                 const float* __restrict__ last_update,
                                 const float* __restrict__ t,
                                 const float* __restrict__ msg,
                                 const float* __restrict__ tw,
                                 const float* __restrict__ tb,
                                 __bf16* __restrict__ eab) {
    size_t idx = (size_t)blockIdx.x * blockDim.x + threadIdx.x;
    if (idx >= (size_t)NE * ED_D) return;
    int e = (int)(idx >> 7);        // / 128
    int c = (int)(idx & 127);
    float v;
    if (c < TD_D) {
        long long s = eidx[e];                      // row 0 = src
        float rel = last_update[s] - t[e];
        v = cosf(rel * tw[c] + tb[c]);
    } else {
        v = msg[(size_t)e * MSG_D + (c - MSG_D)];
    }
    eab[idx] = (__bf16)v;
}

// ---------------------------------------------------------------------------
// Weight-stationary bf16 WMMA GEMM:  C(MxNC) = A(MxK) @ B(KxNC) + bias.
// The whole B matrix is staged once per block into LDS, TRANSPOSED
// (BT[n][k], row stride K+8 elements -> 16B-aligned rows), so each lane's
// B fragment is two contiguous ds_load_b128. A fragments are two 16B global
// loads. Each wave owns two adjacent N-tiles per M-tile (A-fragment reuse,
// back-to-back v_wmma issue), block grid-strides over M tiles.
// Fragment k order is contiguous (k = 16*half + s), the same in A and B.
// ---------------------------------------------------------------------------
template <int K, int NC, typename OutT>
__global__ void gemm_bf16_wmma(const __bf16* __restrict__ A,
                               const __bf16* __restrict__ B,
                               const float*  __restrict__ bias,
                               OutT* __restrict__ C, int M) {
    constexpr int KP = K + 8;                       // padded row stride (elems)
    __shared__ alignas(16) __bf16 BT[NC][KP];

    // ---- stage B -> LDS transposed (coalesced b128 global reads) ----
    constexpr int NVEC = NC / 8;                    // 8 bf16 per b128
    for (int i = threadIdx.x; i < K * NVEC; i += blockDim.x) {
        int k  = i / NVEC;
        int n0 = (i - k * NVEC) * 8;
        union { uint4 u4; __bf16 e[8]; } tmp;
        tmp.u4 = *(const uint4*)(B + (size_t)k * NC + n0);
#pragma unroll
        for (int j = 0; j < 8; ++j) BT[n0 + j][k] = tmp.e[j];
    }
    __syncthreads();

    // ---- wave -> (mSub, nPair) mapping ----
    const int lane = threadIdx.x & 31;
    const int w    = threadIdx.x >> 5;              // 0..7
    constexpr int tilesN      = NC / 16;
    constexpr int nPairs      = tilesN / 2;         // waves needed per M tile
    constexpr int mSubPerBlk  = 8 / nPairs;         // M tiles per block pass
    const int mSub  = w / nPairs;
    const int nPair = w - mSub * nPairs;
    const int halfSel = lane >> 4;                  // 0 | 1
    const int l15     = lane & 15;
    const int col0 = (nPair * 2) * 16 + l15;
    const int col1 = (nPair * 2 + 1) * 16 + l15;
    const int koB  = halfSel * 16;                  // B K sub-offset
    const int khA  = halfSel * 16;                  // A K sub-offset

    const float b0 = bias ? bias[col0] : 0.0f;
    const float b1 = bias ? bias[col1] : 0.0f;
    const int tilesM = M >> 4;

    for (int tmBase = blockIdx.x * mSubPerBlk; tmBase < tilesM;
         tmBase += gridDim.x * mSubPerBlk) {
        const int tm = tmBase + mSub;
        if (tm >= tilesM) continue;                 // wave-uniform

        v8f acc0, acc1;
#pragma unroll
        for (int r = 0; r < 8; ++r) { acc0[r] = b0; acc1[r] = b1; }

        const __bf16* Arow = A + (size_t)(tm * 16 + l15) * K;
#pragma unroll
        for (int kb = 0; kb < K; kb += 32) {
            union { v16bf v; uint4 q[2]; } a;
            a.q[0] = *(const uint4*)(Arow + kb + khA);        // K kh..kh+7
            a.q[1] = *(const uint4*)(Arow + kb + khA + 8);    // K kh+8..kh+15
            union { v16bf v; uint4 q[2]; } bf0, bf1;
            const uint4* p0 = (const uint4*)(&BT[col0][kb + koB]);
            const uint4* p1 = (const uint4*)(&BT[col1][kb + koB]);
            bf0.q[0] = p0[0]; bf0.q[1] = p0[1];
            bf1.q[0] = p1[0]; bf1.q[1] = p1[1];
            acc0 = __builtin_amdgcn_wmma_f32_16x16x32_bf16(
                false, a.v, false, bf0.v, (short)0, acc0, false, false);
            acc1 = __builtin_amdgcn_wmma_f32_16x16x32_bf16(
                false, a.v, false, bf1.v, (short)0, acc1, false, false);
        }

        // ---- epilogue: one base address, unroll-constant row offsets so the
        // backend folds r*NC into the store instruction's immediate offset.
        const int rbase = tm * 16 + halfSel * 8;    // D: VGPR r -> M=rbase+r
        OutT* cp0 = C + (size_t)rbase * NC + col0;
        OutT* cp1 = C + (size_t)rbase * NC + col1;
#pragma unroll
        for (int r = 0; r < 8; ++r) {
            cp0[r * NC] = (OutT)acc0[r];
            cp1[r * NC] = (OutT)acc1[r];
        }
    }
}

// ---------------------------------------------------------------------------
// Segment-softmax attention (3 global passes; kernel-launch boundaries give
// the required ordering).
// ---------------------------------------------------------------------------
template <int HEADS>
__global__ void attn_logits_kernel(const long long* __restrict__ eidx,
                                   const float* __restrict__ Q,
                                   const float* __restrict__ Km,
                                   const __bf16* __restrict__ Eb,
                                   float* __restrict__ logits,
                                   float* __restrict__ mx) {
    size_t idx = (size_t)blockIdx.x * blockDim.x + threadIdx.x;
    if (idx >= (size_t)NE * HEADS) return;
    int e = (int)(idx / HEADS);
    int h = (int)(idx - (size_t)e * HEADS);
    long long s = eidx[e];
    long long d = eidx[(size_t)NE + e];
    const float4* q4 = (const float4*)(Q  + (size_t)d * (HEADS * 32) + h * 32);
    const float4* k4 = (const float4*)(Km + (size_t)s * (HEADS * 32) + h * 32);
    const uint2*  e4 = (const uint2*) (Eb + (size_t)e * (HEADS * 32) + h * 32);
    float acc = 0.0f;
#pragma unroll
    for (int j = 0; j < 8; ++j) {
        float4 qv = q4[j];
        float4 kv = k4[j];
        uint2  eu = e4[j];
        float2 e01 = bf16x2_to_f32x2(eu.x);
        float2 e23 = bf16x2_to_f32x2(eu.y);
        acc += qv.x * (kv.x + e01.x) + qv.y * (kv.y + e01.y)
             + qv.z * (kv.z + e23.x) + qv.w * (kv.w + e23.y);
    }
    float lg = acc * 0.17677669529663687f;          // 1/sqrt(32)
    logits[idx] = lg;
    atomicMaxF(&mx[(size_t)d * HEADS + h], lg);
}

template <int HEADS>
__global__ void attn_exp_kernel(const long long* __restrict__ eidx,
                                float* __restrict__ logits,
                                const float* __restrict__ mx,
                                float* __restrict__ sum) {
    size_t idx = (size_t)blockIdx.x * blockDim.x + threadIdx.x;
    if (idx >= (size_t)NE * HEADS) return;
    int e = (int)(idx / HEADS);
    int h = (int)(idx - (size_t)e * HEADS);
    long long d = eidx[(size_t)NE + e];
    float m = mx[(size_t)d * HEADS + h];
    if (!(m > -1e38f)) m = 0.0f;                    // match ref: -inf -> 0
    float ex = __expf(logits[idx] - m);
    logits[idx] = ex;
    atomicAdd(&sum[(size_t)d * HEADS + h], ex);
}

// One wave per edge: lanes cover the 32 channels of each head, so v/e gathers
// and the atomic scatter are wave-coalesced 128B transactions.
template <int HEADS>
__global__ void attn_accum_kernel(const long long* __restrict__ eidx,
                                  const float* __restrict__ logits,
                                  const float* __restrict__ sum,
                                  const float* __restrict__ V,
                                  const __bf16* __restrict__ Eb,
                                  float* __restrict__ out) {
    const int lane = threadIdx.x & 31;
    const int e    = blockIdx.x * (blockDim.x >> 5) + (threadIdx.x >> 5);
    if (e >= NE) return;
    long long s = eidx[e];
    long long d = eidx[(size_t)NE + e];
    const float*  vbase = V   + (size_t)s * (HEADS * 32);
    const __bf16* ebase = Eb  + (size_t)e * (HEADS * 32);
    float*        obase = out + (size_t)d * (HEADS * 32);
#pragma unroll
    for (int h = 0; h < HEADS; ++h) {
        float coeff = logits[(size_t)e * HEADS + h] /
                      (sum[(size_t)d * HEADS + h] + 1e-16f);
        float vv = vbase[h * 32 + lane];
        float ee = (float)ebase[h * 32 + lane];
        atomicAdd(&obase[h * 32 + lane], coeff * (vv + ee));
    }
}

__global__ void relu_to_bf16_kernel(const float* __restrict__ in,
                                    __bf16* __restrict__ out, size_t n) {
    size_t i = (size_t)blockIdx.x * blockDim.x + threadIdx.x;
    if (i < n) out[i] = (__bf16)fmaxf(in[i], 0.0f);
}

__global__ void relu_f32_kernel(const float* __restrict__ in,
                                float* __restrict__ out, size_t n) {
    size_t i = (size_t)blockIdx.x * blockDim.x + threadIdx.x;
    if (i < n) out[i] = fmaxf(in[i], 0.0f);
}

// ---------------------------------------------------------------------------
// Launch
// ---------------------------------------------------------------------------
extern "C" void kernel_launch(void* const* d_in, const int* in_sizes, int n_in,
                              void* d_out, int out_size, void* d_ws, size_t ws_size,
                              hipStream_t stream) {
    // ---- inputs (setup_inputs dict order) ----
    const float*     x    = (const float*)d_in[0];
    const float*     lu   = (const float*)d_in[1];
    const long long* eidx = (const long long*)d_in[2];   // (2,E) int64
    const float*     t    = (const float*)d_in[3];
    const float*     msg  = (const float*)d_in[4];
    const float*     tw   = (const float*)d_in[5];
    const float*     tb   = (const float*)d_in[6];
    const float *Wq1 = (const float*)d_in[7],  *bq1 = (const float*)d_in[8];
    const float *Wk1 = (const float*)d_in[9],  *bk1 = (const float*)d_in[10];
    const float *Wv1 = (const float*)d_in[11], *bv1 = (const float*)d_in[12];
    const float *We1 = (const float*)d_in[13];
    const float *Ws1 = (const float*)d_in[14], *bs1 = (const float*)d_in[15];
    const float *Wq2 = (const float*)d_in[16], *bq2 = (const float*)d_in[17];
    const float *Wk2 = (const float*)d_in[18], *bk2 = (const float*)d_in[19];
    const float *Wv2 = (const float*)d_in[20], *bv2 = (const float*)d_in[21];
    const float *We2 = (const float*)d_in[22];
    const float *Ws2 = (const float*)d_in[23], *bs2 = (const float*)d_in[24];

    // ---- workspace arena ----
    char*  ws  = (char*)d_ws;
    size_t off = 0;
    auto alloc = [&](size_t bytes) -> void* {
        void* p = ws + off;
        off = (off + bytes + 255) & ~(size_t)255;
        return p;
    };
    __bf16* XB   = (__bf16*)alloc((size_t)NN * IN_CH * 2);
    __bf16* EAB  = (__bf16*)alloc((size_t)NE * ED_D * 2);
    __bf16* Wq1b = (__bf16*)alloc((size_t)IN_CH * HID_D * 2);
    __bf16* Wk1b = (__bf16*)alloc((size_t)IN_CH * HID_D * 2);
    __bf16* Wv1b = (__bf16*)alloc((size_t)IN_CH * HID_D * 2);
    __bf16* Ws1b = (__bf16*)alloc((size_t)IN_CH * HID_D * 2);
    __bf16* We1b = (__bf16*)alloc((size_t)ED_D * HID_D * 2);
    __bf16* Wq2b = (__bf16*)alloc((size_t)HID_D * C_CH * 2);
    __bf16* Wk2b = (__bf16*)alloc((size_t)HID_D * C_CH * 2);
    __bf16* Wv2b = (__bf16*)alloc((size_t)HID_D * C_CH * 2);
    __bf16* Ws2b = (__bf16*)alloc((size_t)HID_D * C_CH * 2);
    __bf16* We2b = (__bf16*)alloc((size_t)ED_D * C_CH * 2);
    float*  Q1   = (float*)alloc((size_t)NN * HID_D * 4);
    float*  K1   = (float*)alloc((size_t)NN * HID_D * 4);
    float*  V1   = (float*)alloc((size_t)NN * HID_D * 4);
    float*  S1   = (float*)alloc((size_t)NN * HID_D * 4);   // skip+bias base
    __bf16* E1B  = (__bf16*)alloc((size_t)NE * HID_D * 2);
    float*  LOG1 = (float*)alloc((size_t)NE * H1_H * 4);
    float*  MAX1 = (float*)alloc((size_t)NN * H1_H * 4);
    float*  SUM1 = (float*)alloc((size_t)NN * H1_H * 4);
    __bf16* H1B  = (__bf16*)alloc((size_t)NN * HID_D * 2);
    float*  Q2   = (float*)alloc((size_t)NN * C_CH * 4);
    float*  K2   = (float*)alloc((size_t)NN * C_CH * 4);
    float*  V2   = (float*)alloc((size_t)NN * C_CH * 4);
    float*  S2   = (float*)alloc((size_t)NN * C_CH * 4);
    __bf16* E2B  = (__bf16*)alloc((size_t)NE * C_CH * 2);
    float*  LOG2 = (float*)alloc((size_t)NE * 4);
    float*  MAX2 = (float*)alloc((size_t)NN * 4);
    float*  SUM2 = (float*)alloc((size_t)NN * 4);

    const int TPB = 256;
    auto cdiv = [](size_t a, size_t b) -> unsigned { return (unsigned)((a + b - 1) / b); };

    // ---- 1. fp32 -> bf16 conversions ----
    auto cvt = [&](const float* src, __bf16* dst, size_t n) {
        f32_to_bf16_kernel<<<cdiv(n, TPB), TPB, 0, stream>>>(src, dst, n);
    };
    cvt(x,   XB,   (size_t)NN * IN_CH);
    cvt(Wq1, Wq1b, (size_t)IN_CH * HID_D);
    cvt(Wk1, Wk1b, (size_t)IN_CH * HID_D);
    cvt(Wv1, Wv1b, (size_t)IN_CH * HID_D);
    cvt(Ws1, Ws1b, (size_t)IN_CH * HID_D);
    cvt(We1, We1b, (size_t)ED_D * HID_D);
    cvt(Wq2, Wq2b, (size_t)HID_D * C_CH);
    cvt(Wk2, Wk2b, (size_t)HID_D * C_CH);
    cvt(Wv2, Wv2b, (size_t)HID_D * C_CH);
    cvt(Ws2, Ws2b, (size_t)HID_D * C_CH);
    cvt(We2, We2b, (size_t)ED_D * C_CH);

    // ---- 2. edge_attr (bf16) ----
    edge_attr_kernel<<<cdiv((size_t)NE * ED_D, TPB), TPB, 0, stream>>>(
        eidx, lu, t, msg, tw, tb, EAB);

    // ---- 3. layer-1 GEMMs (WMMA, weight-stationary in LDS) ----
    auto gblocks = [&](int M, int mSubPerBlk) -> unsigned {
        size_t groups = ((size_t)(M / 16) + mSubPerBlk - 1) / mSubPerBlk;
        return (unsigned)(groups < 4096 ? groups : 4096);
    };
    // <K=128, NC=256>: nPairs=8 -> 1 M-tile per block pass
    gemm_bf16_wmma<128, 256, float><<<gblocks(NN, 1), 256, 0, stream>>>(
        XB, Wq1b, bq1, Q1, NN);
    gemm_bf16_wmma<128, 256, float><<<gblocks(NN, 1), 256, 0, stream>>>(
        XB, Wk1b, bk1, K1, NN);
    gemm_bf16_wmma<128, 256, float><<<gblocks(NN, 1), 256, 0, stream>>>(
        XB, Wv1b, bv1, V1, NN);
    gemm_bf16_wmma<128, 256, float><<<gblocks(NN, 1), 256, 0, stream>>>(
        XB, Ws1b, bs1, S1, NN);
    gemm_bf16_wmma<128, 256, __bf16><<<gblocks(NE, 1), 256, 0, stream>>>(
        EAB, We1b, nullptr, E1B, NE);

    // ---- 4. layer-1 segment softmax ----
    fill_f32_kernel<<<cdiv((size_t)NN * H1_H, TPB), TPB, 0, stream>>>(
        MAX1, -__builtin_inff(), (size_t)NN * H1_H);
    fill_f32_kernel<<<cdiv((size_t)NN * H1_H, TPB), TPB, 0, stream>>>(
        SUM1, 0.0f, (size_t)NN * H1_H);
    attn_logits_kernel<H1_H><<<cdiv((size_t)NE * H1_H, TPB), TPB, 0, stream>>>(
        eidx, Q1, K1, E1B, LOG1, MAX1);
    attn_exp_kernel<H1_H><<<cdiv((size_t)NE * H1_H, TPB), TPB, 0, stream>>>(
        eidx, LOG1, MAX1, SUM1);
    attn_accum_kernel<H1_H><<<cdiv((size_t)NE * 32, TPB), TPB, 0, stream>>>(
        eidx, LOG1, SUM1, V1, E1B, S1);

    // ---- 5. h = relu(attn + skip) -> bf16 ----
    relu_to_bf16_kernel<<<cdiv((size_t)NN * HID_D, TPB), TPB, 0, stream>>>(
        S1, H1B, (size_t)NN * HID_D);

    // ---- 6. layer-2 GEMMs (WMMA) ----
    // <K=256, NC=32>: nPairs=1 -> 8 M-tiles per block pass
    gemm_bf16_wmma<256, 32, float><<<gblocks(NN, 8), 256, 0, stream>>>(
        H1B, Wq2b, bq2, Q2, NN);
    gemm_bf16_wmma<256, 32, float><<<gblocks(NN, 8), 256, 0, stream>>>(
        H1B, Wk2b, bk2, K2, NN);
    gemm_bf16_wmma<256, 32, float><<<gblocks(NN, 8), 256, 0, stream>>>(
        H1B, Wv2b, bv2, V2, NN);
    gemm_bf16_wmma<256, 32, float><<<gblocks(NN, 8), 256, 0, stream>>>(
        H1B, Ws2b, bs2, S2, NN);
    gemm_bf16_wmma<128, 32, __bf16><<<gblocks(NE, 8), 256, 0, stream>>>(
        EAB, We2b, nullptr, E2B, NE);

    // ---- 7. layer-2 segment softmax (1 head) ----
    fill_f32_kernel<<<cdiv((size_t)NN, TPB), TPB, 0, stream>>>(
        MAX2, -__builtin_inff(), (size_t)NN);
    fill_f32_kernel<<<cdiv((size_t)NN, TPB), TPB, 0, stream>>>(
        SUM2, 0.0f, (size_t)NN);
    attn_logits_kernel<1><<<cdiv((size_t)NE, TPB), TPB, 0, stream>>>(
        eidx, Q2, K2, E2B, LOG2, MAX2);
    attn_exp_kernel<1><<<cdiv((size_t)NE, TPB), TPB, 0, stream>>>(
        eidx, LOG2, MAX2, SUM2);
    attn_accum_kernel<1><<<cdiv((size_t)NE * 32, TPB), TPB, 0, stream>>>(
        eidx, LOG2, SUM2, V2, E2B, S2);

    // ---- 8. out = relu(attn + skip) ----
    relu_f32_kernel<<<cdiv((size_t)NN * C_CH, TPB), TPB, 0, stream>>>(
        S2, (float*)d_out, (size_t)NN * C_CH);
}
